// MambaModel_37151467110552
// MI455X (gfx1250) — matrix operand order repaired and checked by reading, not compiled, not run on hardware
//
#include <hip/hip_runtime.h>
#include <hip/hip_fp16.h>

typedef __attribute__((ext_vector_type(16))) _Float16 v16h;
typedef __attribute__((ext_vector_type(8)))  float    v8f;

namespace mk {

constexpr int NL = 12;      // layers
constexpr int H  = 768;     // model dim
constexpr int DI = 1536;    // inner dim
constexpr int SN = 16;      // state dim N
constexpr int R  = 48;      // dt rank
constexpr int RP = 64;      // dt rank padded to K multiple of 32
constexpr int KC = 4;       // conv kernel width
constexpr int V  = 1024;    // vocab
constexpr int L  = 1024;    // sequence length
constexpr int B  = 4;       // batch
constexpr int MT = L * B;   // 4096 tokens
constexpr int P  = R + 2 * SN; // 80 = x_proj output dim

// ---------------------------------------------------------------- utilities

__global__ __launch_bounds__(256)
void cvt_f32_to_f16(const float* __restrict__ in, _Float16* __restrict__ out, int n) {
  int i = blockIdx.x * 256 + threadIdx.x;
  if (i < n) out[i] = (_Float16)in[i];
}

// dt_w (NL*DI, R) -> f16 rows padded to stride RP with zeros
__global__ __launch_bounds__(256)
void pad_dtw_kernel(const float* __restrict__ in, _Float16* __restrict__ o) {
  int i = blockIdx.x * 256 + threadIdx.x;      // NL*DI*RP total
  if (i >= NL * DI * RP) return;
  int r = i / RP, c = i - r * RP;
  o[i] = (c < R) ? (_Float16)in[r * R + c] : (_Float16)0.f;
}

// h[b*L+l][c] = embed[src[l][b]][c]   (ids = src.T)
__global__ __launch_bounds__(256)
void embed_kernel(const int* __restrict__ src, const float* __restrict__ embed,
                  float* __restrict__ h) {
  int idx = blockIdx.x * 256 + threadIdx.x;      // MT*H total
  if (idx >= MT * H) return;
  int m = idx / H;
  int c = idx - m * H;
  int b = m / L, l = m - b * L;
  int tok = src[l * B + b];
  h[idx] = embed[(size_t)tok * H + c];
}

// out(f16) = x * rsqrt(mean(x^2)+eps) * w, one block (8 waves) per token row
__global__ __launch_bounds__(256)
void rmsnorm_f16_kernel(const float* __restrict__ x, const float* __restrict__ w,
                        _Float16* __restrict__ out) {
  __shared__ float red[9];
  int m = blockIdx.x;
  const float* row = x + (size_t)m * H;
  float ss = 0.f;
  for (int c = threadIdx.x; c < H; c += 256) { float v = row[c]; ss += v * v; }
  for (int off = 16; off > 0; off >>= 1) ss += __shfl_down(ss, off, 32);
  if ((threadIdx.x & 31) == 0) red[threadIdx.x >> 5] = ss;
  __syncthreads();
  if (threadIdx.x == 0) {
    float t = 0.f;
    for (int i = 0; i < 8; ++i) t += red[i];
    red[8] = rsqrtf(t / (float)H + 1e-5f);
  }
  __syncthreads();
  float scale = red[8];
  for (int c = threadIdx.x; c < H; c += 256)
    out[(size_t)m * H + c] = (_Float16)(row[c] * scale * w[c]);
}

// causal depthwise conv (K=4) + bias + SiLU; reads xc = xz[:, :DI]
__global__ __launch_bounds__(256)
void conv_silu_kernel(const _Float16* __restrict__ xz,  // (MT, 2*DI) f16
                      const float* __restrict__ cw,     // (DI, KC)
                      const float* __restrict__ cb,     // (DI)
                      _Float16* __restrict__ conv_h) {  // (MT, DI) f16
  int idx = blockIdx.x * 256 + threadIdx.x;             // MT*DI total
  if (idx >= MT * DI) return;
  int m = idx / DI;
  int d = idx - m * DI;
  int b = m / L, l = m - b * L;
  float acc = cb[d];
#pragma unroll
  for (int k = 0; k < KC; ++k) {
    int ls = l - (KC - 1) + k;
    if (ls >= 0)
      acc += (float)xz[(size_t)(b * L + ls) * (2 * DI) + d] * cw[d * KC + k];
  }
  acc = acc / (1.f + __expf(-acc));                     // SiLU
  conv_h[idx] = (_Float16)acc;
}

// selective scan: one thread per (b, d); state s[SN] in registers; fused
// y = (scan + x*D) * silu(z), written as f16 A-matrix for out_proj GEMM.
__global__ __launch_bounds__(128)
void scan_kernel(const _Float16* __restrict__ dt_h,   // (MT, DI)
                 const _Float16* __restrict__ conv_h, // (MT, DI)
                 const float* __restrict__ proj,      // (MT, P)  [dt_r | B | C]
                 const _Float16* __restrict__ xz,     // (MT, 2*DI) for z
                 const float* __restrict__ A_log,     // (DI, SN) this layer
                 const float* __restrict__ Dw,        // (DI) this layer
                 _Float16* __restrict__ y_h) {        // (MT, DI)
  int t = blockIdx.x * 128 + threadIdx.x;
  if (t >= B * DI) return;
  int b = t / DI, d = t - b * DI;
  float Ar[SN], s[SN];
#pragma unroll
  for (int n = 0; n < SN; ++n) {
    Ar[n] = -__expf(A_log[d * SN + n]);
    s[n]  = 0.f;
  }
  const float Dv = Dw[d];
  for (int l = 0; l < L; ++l) {
    size_t m = (size_t)b * L + l;
    float dtv = (float)dt_h[m * DI + d];
    float xv  = (float)conv_h[m * DI + d];
    const float* pr = proj + m * P;
    float du = dtv * xv;
    float y = 0.f;
#pragma unroll
    for (int n = 0; n < SN; ++n) {
      float dA = __expf(dtv * Ar[n]);
      s[n] = s[n] * dA + du * pr[R + n];        // B_t
      y += s[n] * pr[R + SN + n];               // C_t
    }
    y += xv * Dv;
    float z = (float)xz[m * (2 * DI) + DI + d];
    y *= z / (1.f + __expf(-z));                // * silu(z)
    y_h[m * DI + d] = (_Float16)y;
  }
}

// ------------------------------------------------------------- WMMA GEMM
// D(M,N) = A(M,K) x W(N,K)^T, f16 inputs, fp32 WMMA accumulation.
// B tile is staged through LDS with CDNA5 async copies
// (global_load_async_to_lds_b128 / ASYNCcnt), double-buffered: the async
// engine fills buffer p^1 for the next K-step while WMMAs consume buffer p
// via ds_load_b128. One s_wait_asynccnt + workgroup barrier per K-step.
// A stays on the global->register path (per-wave, double-buffered).
// Fragment layouts:
//   A (16-bit 16x32): lanes 0-15 row=m, K kk..kk+15;
//                     lanes 16-31 row=m, K kk+16..kk+31.
//   B (32x16): lane -> col n; lanes 0-15 K kk..kk+15, lanes 16-31 K kk+16..31;
//              LDS holds 32 contiguous f16 of W row n at byte n*64.
//   C: vgpr i -> row (hi*8+i), col = lane&15.

enum { EP_F16 = 0, EP_PROJ = 1, EP_DT = 2, EP_RESID = 3, EP_LOGITS = 4 };

__device__ __forceinline__ v16h load_frag_g(const _Float16* __restrict__ p) {
  union { v16h v; uint4 q[2]; } u;
  u.q[0] = *(const uint4*)(p);
  u.q[1] = *(const uint4*)(p + 8);
  return u.v;
}

template <int WM, int WN, int EP, int N, int K, int WAVES>
__global__ __launch_bounds__(32 * WAVES)
void gemm_wmma(const _Float16* __restrict__ A,   // (M, K)
               const _Float16* __restrict__ Bw,  // (N, K)
               _Float16* __restrict__ Cf16,
               float* __restrict__ Cf32,
               const float* __restrict__ bias,   // EP_DT
               float* __restrict__ resid,        // EP_RESID (in/out fp32)
               _Float16* __restrict__ dtr_h) {   // EP_PROJ secondary output
  static_assert(K % 32 == 0, "K must be a multiple of 32");
  static_assert(N % (16 * WN * WAVES) == 0, "N must tile exactly");

  constexpr int BT  = 16 * WN * WAVES;        // block N-tile (W rows per K-step)
  constexpr int CH  = BT * 4;                 // 16-byte chunks per K-step tile
  constexpr int OPS = CH / (32 * WAVES);      // async copies per thread
  static_assert(CH % (32 * WAVES) == 0, "async chunks must divide threads");

  __shared__ __align__(16) _Float16 ldsB[2][BT * 32];

  const int tid  = threadIdx.x;
  const int lane = tid & 31;
  const int wave = tid >> 5;
  const int row  = lane & 15;
  const int hi   = lane >> 4;
  const int tm   = blockIdx.x * (16 * WM);
  const int bn0  = blockIdx.y * BT;

  // async-copy next B tile (BT rows x 64B) into LDS buffer `buf`
  auto issueB = [&](int kk, int buf) {
#pragma unroll
    for (int o = 0; o < OPS; ++o) {
      int chunk = o * (32 * WAVES) + tid;
      int r = chunk >> 2;
      int part = (chunk & 3) * 16;  // byte offset within the 64B row-slice
      const char* g = (const char*)(Bw + (size_t)(bn0 + r) * K + kk) + part;
      unsigned int l = (unsigned int)(size_t)(&ldsB[buf][0]) + (unsigned int)(r * 64 + part);
      asm volatile("global_load_async_to_lds_b128 %0, %1, off"
                   :: "v"(l), "v"(g)
                   : "memory");
    }
  };

  v8f acc[WM][WN];
#pragma unroll
  for (int w = 0; w < WM; ++w)
#pragma unroll
    for (int j = 0; j < WN; ++j)
#pragma unroll
      for (int i = 0; i < 8; ++i) acc[w][j][i] = 0.f;

  const _Float16* ap[WM];
#pragma unroll
  for (int w = 0; w < WM; ++w)
    ap[w] = A + (size_t)(tm + w * 16 + row) * K + hi * 16;

  issueB(0, 0);                         // prime LDS pipeline
  v16h a_cur[WM], a_nxt[WM];
#pragma unroll
  for (int w = 0; w < WM; ++w) a_cur[w] = load_frag_g(ap[w]);

#pragma unroll
  for (int kk = 0; kk < K; kk += 32) {
    const int buf = (kk >> 5) & 1;
    asm volatile("s_wait_asynccnt 0x0" ::: "memory");  // this buffer landed
    __syncthreads();                                   // visible to all waves
    if (kk + 32 < K) {
      issueB(kk + 32, buf ^ 1);                        // overlap next fill
#pragma unroll
      for (int w = 0; w < WM; ++w) a_nxt[w] = load_frag_g(ap[w] + kk + 32);
    }
#pragma unroll
    for (int j = 0; j < WN; ++j) {
      const int nl = (wave * WN + j) * 16 + row;       // W row within tile
      union { v16h v; uint4 q[2]; } bf;
      const uint4* pl = (const uint4*)&ldsB[buf][nl * 32 + hi * 16];
      bf.q[0] = pl[0];                                 // ds_load_b128
      bf.q[1] = pl[1];                                 // ds_load_b128
#pragma unroll
      for (int w = 0; w < WM; ++w)
        acc[w][j] = __builtin_amdgcn_wmma_f32_16x16x32_f16(
            false, a_cur[w], false, bf.v, (short)0, acc[w][j], false, false);
    }
    if (kk + 32 < K) {
#pragma unroll
      for (int w = 0; w < WM; ++w) a_cur[w] = a_nxt[w];
    }
  }

  const int tn0 = bn0 + wave * (16 * WN);
#pragma unroll
  for (int w = 0; w < WM; ++w) {
#pragma unroll
    for (int j = 0; j < WN; ++j) {
#pragma unroll
      for (int i = 0; i < 8; ++i) {
        const int m = tm + w * 16 + hi * 8 + i;
        const int n = tn0 + j * 16 + row;
        const float c = acc[w][j][i];
        if (EP == EP_F16) {
          Cf16[(size_t)m * N + n] = (_Float16)c;
        } else if (EP == EP_PROJ) {
          Cf32[(size_t)m * N + n] = c;
          // dt_r slice, stored padded to RP=64 so the dt GEMM has K%32==0
          if (n < R)       dtr_h[(size_t)m * RP + n] = (_Float16)c;
          else if (n < RP) dtr_h[(size_t)m * RP + n] = (_Float16)0.f;
        } else if (EP == EP_DT) {
          float s = c + bias[n];
          s = (s > 20.f) ? s : log1pf(__expf(s));              // softplus
          Cf16[(size_t)m * N + n] = (_Float16)s;
        } else if (EP == EP_RESID) {
          const size_t o = (size_t)m * N + n;
          resid[o] = resid[o] + c;                             // residual add
        } else {  // EP_LOGITS: (B,L,V) -> (L,B,V)
          const int b = m / L, l = m - b * L;
          Cf32[((size_t)l * B + b) * N + n] = c;
        }
      }
    }
  }
}

}  // namespace mk

// ------------------------------------------------------------------- host

extern "C" void kernel_launch(void* const* d_in, const int* in_sizes, int n_in,
                              void* d_out, int out_size, void* d_ws, size_t ws_size,
                              hipStream_t stream) {
  using namespace mk;
  const int*   src        = (const int*)  d_in[0];
  const float* embed_w    = (const float*)d_in[1];
  const float* norm_w     = (const float*)d_in[2];
  const float* in_proj_w  = (const float*)d_in[3];
  const float* conv_w     = (const float*)d_in[4];
  const float* conv_b     = (const float*)d_in[5];
  const float* x_proj_w   = (const float*)d_in[6];
  const float* dt_w       = (const float*)d_in[7];
  const float* dt_b       = (const float*)d_in[8];
  const float* A_log      = (const float*)d_in[9];
  const float* Dw         = (const float*)d_in[10];
  const float* out_proj_w = (const float*)d_in[11];
  const float* norm_f_w   = (const float*)d_in[12];
  const float* lm_head_w  = (const float*)d_in[13];
  float* out = (float*)d_out;

  char* ws = (char*)d_ws;
  size_t off = 0;
  auto carve = [&](size_t bytes) -> void* {
    void* p = ws + off;
    off = (off + bytes + 255) & ~(size_t)255;
    return p;
  };

  // activations
  float*    h_buf  = (float*)   carve((size_t)MT * H * 4);       // residual stream
  _Float16* hn_h   = (_Float16*)carve((size_t)MT * H * 2);       // rmsnorm out
  _Float16* xz_h   = (_Float16*)carve((size_t)MT * 2 * DI * 2);  // in_proj out
  _Float16* conv_h = (_Float16*)carve((size_t)MT * DI * 2);      // conv+silu out
  float*    proj_f = (float*)   carve((size_t)MT * P * 4);       // x_proj out
  _Float16* dtr_h  = (_Float16*)carve((size_t)MT * RP * 2);      // dt_r (f16, K-padded)
  _Float16* dt_h   = (_Float16*)carve((size_t)MT * DI * 2);      // softplus(dt)
  _Float16* y_h    = (_Float16*)carve((size_t)MT * DI * 2);      // gated scan out
  // f16 weight copies
  _Float16* ipw_h  = (_Float16*)carve((size_t)NL * 2 * DI * H * 2);
  _Float16* xpw_h  = (_Float16*)carve((size_t)NL * P * DI * 2);
  _Float16* dtw_h  = (_Float16*)carve((size_t)NL * DI * RP * 2); // K-padded
  _Float16* opw_h  = (_Float16*)carve((size_t)NL * H * DI * 2);
  _Float16* lmw_h  = (_Float16*)carve((size_t)V * H * 2);

  auto cvt = [&](const float* in, _Float16* o, int n) {
    cvt_f32_to_f16<<<dim3((n + 255) / 256), dim3(256), 0, stream>>>(in, o, n);
  };
  cvt(in_proj_w,  ipw_h, NL * 2 * DI * H);
  cvt(x_proj_w,   xpw_h, NL * P * DI);
  cvt(out_proj_w, opw_h, NL * H * DI);
  cvt(lm_head_w,  lmw_h, V * H);

  pad_dtw_kernel<<<dim3((NL * DI * RP + 255) / 256), dim3(256), 0, stream>>>(
      dt_w, dtw_h);

  embed_kernel<<<dim3((MT * H + 255) / 256), dim3(256), 0, stream>>>(src, embed_w, h_buf);

  for (int layer = 0; layer < NL; ++layer) {
    rmsnorm_f16_kernel<<<dim3(MT), dim3(256), 0, stream>>>(
        h_buf, norm_w + (size_t)layer * H, hn_h);

    // xz = hn @ in_proj^T  (4096 x 768 -> 3072); 32x256 tile per block
    gemm_wmma<2, 4, EP_F16, 2 * DI, H, 4>
        <<<dim3(MT / 32, (2 * DI) / 256), dim3(128), 0, stream>>>(
        hn_h, ipw_h + (size_t)layer * 2 * DI * H,
        xz_h, nullptr, nullptr, nullptr, nullptr);

    conv_silu_kernel<<<dim3((MT * DI + 255) / 256), dim3(256), 0, stream>>>(
        xz_h, conv_w + (size_t)layer * DI * KC, conv_b + (size_t)layer * DI, conv_h);

    // proj = conv @ x_proj^T  (4096 x 1536 -> 80); writes padded dt_r slice
    gemm_wmma<2, 1, EP_PROJ, P, DI, 5>
        <<<dim3(MT / 32, 1), dim3(160), 0, stream>>>(
        conv_h, xpw_h + (size_t)layer * P * DI,
        nullptr, proj_f, nullptr, nullptr, dtr_h);

    // dt = softplus(dt_r @ dt_w^T + dt_b)  (4096 x 64 -> 1536)
    gemm_wmma<2, 4, EP_DT, DI, RP, 4>
        <<<dim3(MT / 32, DI / 256), dim3(128), 0, stream>>>(
        dtr_h, dtw_h + (size_t)layer * DI * RP,
        dt_h, nullptr, dt_b + (size_t)layer * DI, nullptr, nullptr);

    scan_kernel<<<dim3((B * DI + 127) / 128), dim3(128), 0, stream>>>(
        dt_h, conv_h, proj_f, xz_h,
        A_log + (size_t)layer * DI * SN, Dw + (size_t)layer * DI, y_h);

    // h += y @ out_proj^T  (4096 x 1536 -> 768)
    gemm_wmma<2, 4, EP_RESID, H, DI, 4>
        <<<dim3(MT / 32, H / 256), dim3(128), 0, stream>>>(
        y_h, opw_h + (size_t)layer * H * DI,
        nullptr, nullptr, nullptr, h_buf, nullptr);
  }

  rmsnorm_f16_kernel<<<dim3(MT), dim3(256), 0, stream>>>(h_buf, norm_f_w, hn_h);

  // logits = hn @ lm_head^T  (4096 x 768 -> 1024), transposed store (L,B,V)
  gemm_wmma<2, 4, EP_LOGITS, V, H, 4>
      <<<dim3(MT / 32, V / 256), dim3(128), 0, stream>>>(
      hn_h, lmw_h, nullptr, out, nullptr, nullptr, nullptr);

  (void)in_sizes; (void)n_in; (void)out_size; (void)ws_size;
}
